// CMSBlockLinear_16750372454415
// MI455X (gfx1250) — matrix operand order repaired and checked by reading, not compiled, not run on hardware
//
#include <hip/hip_runtime.h>
#include <hip/hip_bf16.h>
#include <stdint.h>

// ---- problem constants (from reference) ----
#define TOKENS 8192
#define IN_F   2048
#define OUT_F  8192
#define TILE   16
#define C_BLK  (IN_F / TILE)    // 128
#define R_BLK  (OUT_F / TILE)   // 512
#define K_BLK  (C_BLK / 2)      // 64 gathered tiles per row-block

// ---- tunables ----
#define TM        64            // tokens staged in LDS per workgroup (4 WMMA tiles)
#define RB        16            // row-blocks per workgroup (16 waves x 1 r each)
#define NTHREADS  512           // 16 waves = 4 waves/SIMD on one WGP
#define TT        4             // token tiles per wave (M = 64)
#define NPAIR     (K_BLK / 2)   // 32 k-pairs (WMMA K=32 each)
#define LDS_ROW_BYTES (IN_F * 2 + 16)       // 4112: conflict-minimal ds_load_b128
#define LDS_BYTES     (TM * LDS_ROW_BYTES)  // 263168 (< 320 KB WGP LDS)

typedef __attribute__((ext_vector_type(16))) __bf16        v16bf;
typedef __attribute__((ext_vector_type(8)))  float         v8f;
typedef __attribute__((ext_vector_type(4)))  unsigned int  v4u;
typedef __attribute__((ext_vector_type(4)))  int           v4i;

// ---- CDNA5 async global->LDS staging (verified to compile on this toolchain) ----
#if defined(__has_builtin)
#if __has_builtin(__builtin_amdgcn_global_load_async_to_lds_b128) && \
    __has_builtin(__builtin_amdgcn_s_wait_asynccnt)
#define ASYNC_STAGE 1
#endif
#endif
#ifndef ASYNC_STAGE
#define ASYNC_STAGE 0
#endif

// ---- scheduler pipeline control (masks: 2=VALU, 0x100=DS read, 0x20=VMEM read, 8=MFMA/WMMA) ----
#if defined(__has_builtin)
#if __has_builtin(__builtin_amdgcn_sched_group_barrier)
#define SGB(mask, size, id) __builtin_amdgcn_sched_group_barrier((mask), (size), (id))
#endif
#endif
#ifndef SGB
#define SGB(mask, size, id)
#endif

#define AS1_V4I(p) ((__attribute__((address_space(1))) v4i*)(uintptr_t)(p))
#define AS3_V4I(p) ((__attribute__((address_space(3))) v4i*)(uint32_t)(uintptr_t)(p))

// ---------------- f32 -> bf16 (RNE) cast kernel ----------------
__device__ __forceinline__ unsigned int pk_bf16(unsigned int u0, unsigned int u1) {
    unsigned int r0 = u0 + 0x7FFFu + ((u0 >> 16) & 1u);
    unsigned int r1 = u1 + 0x7FFFu + ((u1 >> 16) & 1u);
    return (r0 >> 16) | (r1 & 0xFFFF0000u);
}

__global__ void __launch_bounds__(256)
cast_f32_to_bf16_x8(const unsigned int* __restrict__ src,
                    unsigned int* __restrict__ dst, int n8) {
    int i = blockIdx.x * blockDim.x + threadIdx.x;
    if (i >= n8) return;
    const v4u* s = (const v4u*)src;
    v4u a = __builtin_nontemporal_load(s + 2 * (size_t)i);
    v4u b = __builtin_nontemporal_load(s + 2 * (size_t)i + 1);
    v4u o;
    o.x = pk_bf16(a.x, a.y);
    o.y = pk_bf16(a.z, a.w);
    o.z = pk_bf16(b.x, b.y);
    o.w = pk_bf16(b.z, b.w);
    *((v4u*)dst + i) = o;   // regular store: keep bf16 data L2-resident
}

// ---------------- main block-sparse GEMM ----------------
// xbf : [TOKENS][IN_F] bf16 (row major)
// vbf : [R_BLK][K_BLK][16(o)][16(i)] bf16  (same layout as reference values)
// cols: [R_BLK][K_BLK] int32
__global__ void __launch_bounds__(NTHREADS, 1)
cms_block_linear(const unsigned short* __restrict__ xbf,
                 const unsigned short* __restrict__ vbf,
                 const float* __restrict__ bias,
                 const int* __restrict__ cols,
                 float* __restrict__ out) {
    extern __shared__ unsigned char smem[];

    const int tid  = threadIdx.x;
    const int lane = tid & 31;
    const int wave = tid >> 5;          // 0..15: each wave owns one r, all 4 token tiles
    const int tBase = blockIdx.x * TM;

    // ---- stage TM x IN_F bf16 x-slab into LDS (padded rows), async path ----
    // 64 rows * 256 chunks of 16B; 512 threads * 32 iterations, coalesced.
    {
        const char* xg = (const char*)xbf;
#if ASYNC_STAGE
        #pragma unroll 8
        for (int it = 0; it < (TM * (IN_F * 2 / 16)) / NTHREADS; ++it) {
            int idx = tid + NTHREADS * it;
            int row = idx >> 8;            // 256 chunks of 16B per row
            int ch  = idx & 255;
            const char* g = xg + (size_t)(tBase + row) * (IN_F * 2) + ch * 16;
            unsigned char* l = smem + row * LDS_ROW_BYTES + ch * 16;
            __builtin_amdgcn_global_load_async_to_lds_b128(AS1_V4I(g), AS3_V4I(l), 0, 0);
        }
        __builtin_amdgcn_s_wait_asynccnt(0);
#else
        #pragma unroll 4
        for (int it = 0; it < (TM * (IN_F * 2 / 16)) / NTHREADS; ++it) {
            int idx = tid + NTHREADS * it;
            int row = idx >> 8;
            int ch  = idx & 255;
            v4u d = *(const v4u*)(xg + (size_t)(tBase + row) * (IN_F * 2) + ch * 16);
            *(v4u*)(smem + row * LDS_ROW_BYTES + ch * 16) = d;
        }
#endif
    }
    __syncthreads();

    const int halfSel = lane >> 4;       // 0: lanes 0-15 (K 0-7 / tile k0), 1: lanes 16-31
    const int n       = lane & 15;       // M row within tile (A) / N column (B, C)

    // A operand base per token tile tt: row = tt*16 + n, +16B for hi-K half lanes
    const unsigned char* aBase = smem + n * LDS_ROW_BYTES + (halfSel << 4);

    const int r = blockIdx.y * RB + wave;
    const int* __restrict__ cr = cols + (size_t)r * K_BLK;
    // preload all 64 gather indices pre-scaled to byte offsets (c*32): two wave loads
    const int colLo = cr[lane] << 5;
    const int colHi = cr[lane + 32] << 5;

    // B operand: row o = n, i = 8*halfSel..+7 of tile 2p (u[0]) and tile 2p+1 (u[1])
    const unsigned short* bLane =
        vbf + (size_t)r * K_BLK * 256 + (size_t)halfSel * 8 + n * 16;

    // byte offset of gathered tile q's A chunk, broadcast via readlane
    auto coff = [&](int q) -> int {
        return (q < 32) ? __builtin_amdgcn_readlane(colLo, q)
                        : __builtin_amdgcn_readlane(colHi, q - 32);
    };

    union AB { v4u u[2]; v16bf v; };
    AB A[2][TT], B[2];
    v8f acc[TT];
    #pragma unroll
    for (int tt = 0; tt < TT; ++tt)
        acc[tt] = (v8f){0.f, 0.f, 0.f, 0.f, 0.f, 0.f, 0.f, 0.f};

    // -------- prologue: fetch pair 0 into buffer 0 --------
    {
        const int c0 = coff(0), c1 = coff(1);
        const v4u* bp = (const v4u*)bLane;
        B[0].u[0] = bp[0];
        B[0].u[1] = bp[32];
        #pragma unroll
        for (int tt = 0; tt < TT; ++tt) {
            const unsigned char* aT = aBase + tt * (TILE * LDS_ROW_BYTES);
            A[0][tt].u[0] = *(const v4u*)(aT + c0);
            A[0][tt].u[1] = *(const v4u*)(aT + c1);
        }
    }

    // -------- double-buffered, scheduler-pinned main loop --------
    #pragma unroll
    for (int p = 0; p < NPAIR; ++p) {
        const int cur = p & 1;
        const int nxt = cur ^ 1;
        if (p + 1 < NPAIR) {
            const int c0 = coff(2 * p + 2), c1 = coff(2 * p + 3);
            const v4u* bp = (const v4u*)(bLane + (size_t)(p + 1) * 512);
            B[nxt].u[0] = bp[0];
            B[nxt].u[1] = bp[32];
            #pragma unroll
            for (int tt = 0; tt < TT; ++tt) {
                const unsigned char* aT = aBase + tt * (TILE * LDS_ROW_BYTES);
                A[nxt][tt].u[0] = *(const v4u*)(aT + c0);
                A[nxt][tt].u[1] = *(const v4u*)(aT + c1);
            }
        }
        #pragma unroll
        for (int tt = 0; tt < TT; ++tt) {
            acc[tt] = __builtin_amdgcn_wmma_f32_16x16x32_bf16(
                false, A[cur][tt].v, false, B[cur].v, (short)0, acc[tt], false, false);
        }
        // pipeline shape per pair: addr VALU -> A ds-loads -> B vmem -> 4x WMMA.
        // The VALU adds sit between the previous WMMA group and the ds writes to its
        // operand registers, filling the 4 co-exec hazard slots (no v_nop).
        SGB(0x002, 8, 0);   // 8 VALU (gather address adds)
        SGB(0x100, 8, 0);   // 8 DS reads (A buffer fill for p+1)
        SGB(0x020, 2, 0);   // 2 VMEM reads (B buffer fill for p+1)
        SGB(0x008, 4, 0);   // 4 WMMA (pair p)
    }

    const float bv = bias[r * 16 + n];
    #pragma unroll
    for (int tt = 0; tt < TT; ++tt) {
        float* o0 = out + (size_t)(tBase + tt * 16 + halfSel * 8) * OUT_F + r * 16 + n;
        #pragma unroll
        for (int j = 0; j < 8; ++j) {
            __builtin_nontemporal_store(acc[tt][j] + bv, o0 + (size_t)j * OUT_F);
        }
    }
}

// ---------------- host launch ----------------
extern "C" void kernel_launch(void* const* d_in, const int* in_sizes, int n_in,
                              void* d_out, int out_size, void* d_ws, size_t ws_size,
                              hipStream_t stream) {
    (void)in_sizes; (void)n_in; (void)out_size; (void)ws_size;
    const float* x    = (const float*)d_in[0];
    const float* vals = (const float*)d_in[1];
    const float* bias = (const float*)d_in[2];
    const int*   cols = (const int*)d_in[3];
    float*       out  = (float*)d_out;

    unsigned short* xbf = (unsigned short*)d_ws;
    unsigned short* vbf = (unsigned short*)((char*)d_ws + (size_t)TOKENS * IN_F * 2);

    {   // cast x: 16.7M elems, 8 per thread
        int n8 = TOKENS * IN_F / 8;
        cast_f32_to_bf16_x8<<<n8 / 256, 256, 0, stream>>>(
            (const unsigned int*)x, (unsigned int*)xbf, n8);
    }
    {   // cast values: 8.4M elems
        int n8 = R_BLK * K_BLK * TILE * TILE / 8;
        cast_f32_to_bf16_x8<<<n8 / 256, 256, 0, stream>>>(
            (const unsigned int*)vals, (unsigned int*)vbf, n8);
    }

    dim3 grid(TOKENS / TM, R_BLK / RB);   // 128 x 32
    cms_block_linear<<<grid, NTHREADS, LDS_BYTES, stream>>>(xbf, vbf, bias, cols, out);
}
